// Generator_28724741276275
// MI455X (gfx1250) — compile-verified, matrix operand order checked by reading, not run on hardware
//
#include <hip/hip_runtime.h>
#include <math.h>

typedef __bf16 bf16_t;
typedef __bf16 v16bf __attribute__((ext_vector_type(16)));
typedef __bf16 v8bf  __attribute__((ext_vector_type(8)));
typedef float  v8f   __attribute__((ext_vector_type(8)));

// Problem constants
constexpr int HD    = 128;   // hidden size
constexpr int TLEN  = 512;   // sequence length
constexpr int SDIM  = 2;     // state size
constexpr int EDIM  = 10;    // embed size
constexpr int WAVES = 4;     // waves per block
constexpr int RPW   = 16;    // batch rows per wave
constexpr int NBLK  = 32;    // 32 * 4 * 16 = 2048 = B

// d_ws layout (bf16 element offsets)
constexpr int WS_WH0   = 0;                  // [384][128]
constexpr int WS_WI1   = 49152;              // [384][128]
constexpr int WS_WH1   = 98304;              // [384][128]
constexpr int WS_WI0   = 147456;             // [384][32]  (K padded 12 -> 32 with zeros)
constexpr int WS_WAP   = 159744;             // [16][128]  (rows 2..15 zero; row s = Wa[s][:])
constexpr int WS_TOTAL = 161792;

// ---------------------------------------------------------------------------
// Fragment loader. Per CDNA5 ISA (CDNA5 ISA, 16-bit 16x32 A / 32x16 B layout):
// element j of lane (idx, hlf) holds K = (j/8)*16 + hlf*8 + (j%8).
// A: row = idx (M).  B: row = ntile*16 + idx (weights stored [n][k] row-major).
// Two contiguous 16-byte loads per fragment -> ds_load_b128 / global_load_b128.
// ---------------------------------------------------------------------------
__device__ __forceinline__ v16bf fragLd(const bf16_t* base, int row, int stride,
                                        int kb, int hlf) {
  const bf16_t* p = base + row * stride + kb + hlf * 8;
  union { v16bf v; v8bf h[2]; } u;
  u.h[0] = *(const v8bf*)(p);
  u.h[1] = *(const v8bf*)(p + 16);
  return u.v;
}

__device__ __forceinline__ v8f wmma_bf16(v16bf a, v16bf b, v8f c) {
  return __builtin_amdgcn_wmma_f32_16x16x32_bf16(false, a, false, b, (short)0, c,
                                                 false, false);
}

__device__ __forceinline__ float fsig(float x) {
  return __builtin_amdgcn_rcpf(1.0f + __expf(-x));
}
__device__ __forceinline__ float ftanh(float x) {
  float e = __expf(2.0f * x);
  return (e - 1.0f) * __builtin_amdgcn_rcpf(e + 1.0f);
}

// ---------------------------------------------------------------------------
// One GRU layer for a 16-row batch tile owned by one wave.
//   gates (r,z,n) are 384 columns; processed as 8 groups of 16 columns.
//   r/z: gi+gh accumulate into one C (bias bi+bh pre-folded).
//   n:   separate inn / hn accumulators -> tanh(inn + r*hn).
//   Hlds ([16][128] bf16) updated in place (A-frags already in registers).
// bias layout: [0..255] = bi+bh for r,z; [256..383] = bi_n; [384..511] = bh_n.
// ---------------------------------------------------------------------------
template <int KFI, int WISTRIDE>
__device__ __forceinline__ void gru_layer(int idx, int hlf,
                                          const v16bf* aH, const v16bf* aI,
                                          const bf16_t* __restrict__ Wh,
                                          const bf16_t* __restrict__ Wi,
                                          const float* __restrict__ bias,
                                          bf16_t* __restrict__ Hlds) {
#pragma unroll 1
  for (int g = 0; g < 8; ++g) {
    const int col = g * 16 + idx;
    const float br  = bias[col];
    const float bz  = bias[128 + col];
    const float bni = bias[256 + col];
    const float bnh = bias[384 + col];
    v8f cr = {br, br, br, br, br, br, br, br};
    v8f cz = {bz, bz, bz, bz, bz, bz, bz, bz};
    v8f ci = {bni, bni, bni, bni, bni, bni, bni, bni};
    v8f ch = {bnh, bnh, bnh, bnh, bnh, bnh, bnh, bnh};
#pragma unroll
    for (int k = 0; k < 4; ++k) {  // hidden contribution, K = 128
      const int kb = k * 32;
      cr = wmma_bf16(aH[k], fragLd(Wh, (g     ) * 16 + idx, HD, kb, hlf), cr);
      cz = wmma_bf16(aH[k], fragLd(Wh, (g +  8) * 16 + idx, HD, kb, hlf), cz);
      ch = wmma_bf16(aH[k], fragLd(Wh, (g + 16) * 16 + idx, HD, kb, hlf), ch);
    }
#pragma unroll
    for (int k = 0; k < KFI; ++k) {  // input contribution
      const int kb = k * 32;
      cr = wmma_bf16(aI[k], fragLd(Wi, (g     ) * 16 + idx, WISTRIDE, kb, hlf), cr);
      cz = wmma_bf16(aI[k], fragLd(Wi, (g +  8) * 16 + idx, WISTRIDE, kb, hlf), cz);
      ci = wmma_bf16(aI[k], fragLd(Wi, (g + 16) * 16 + idx, WISTRIDE, kb, hlf), ci);
    }
    // elementwise gate fusion; C layout: elem e -> row m = e + 8*hlf, col = col
#pragma unroll
    for (int e = 0; e < 8; ++e) {
      const int m = e + hlf * 8;
      const float r = fsig(cr[e]);
      const float z = fsig(cz[e]);
      const float n = ftanh(ci[e] + r * ch[e]);
      const float ho = (float)Hlds[m * HD + col];
      Hlds[m * HD + col] = (bf16_t)((1.0f - z) * n + z * ho);
    }
  }
}

// ---------------------------------------------------------------------------
// Weight prep: fp32 -> bf16 into d_ws, padding Wi0 (K 12->32) and Wa (N 2->16).
// ---------------------------------------------------------------------------
__global__ void prep_kernel(const float* __restrict__ Wh0,
                            const float* __restrict__ Wi1,
                            const float* __restrict__ Wh1,
                            const float* __restrict__ Wi0,
                            const float* __restrict__ Wa,
                            bf16_t* __restrict__ ws) {
  int i = blockIdx.x * blockDim.x + threadIdx.x;
  if (i >= WS_TOTAL) return;
  float v;
  if (i < WS_WI1) {
    v = Wh0[i];
  } else if (i < WS_WH1) {
    v = Wi1[i - WS_WI1];
  } else if (i < WS_WI0) {
    v = Wh1[i - WS_WH1];
  } else if (i < WS_WAP) {
    int j = i - WS_WI0, g = j >> 5, k = j & 31;
    v = (k < SDIM + EDIM) ? Wi0[g * (SDIM + EDIM) + k] : 0.0f;
  } else {
    int j = i - WS_WAP, n = j >> 7, k = j & 127;
    v = (n < SDIM) ? Wa[n * HD + k] : 0.0f;
  }
  ws[i] = (bf16_t)v;
}

// ---------------------------------------------------------------------------
// Persistent decoder: each wave owns 16 batch rows for all 512 steps.
// ---------------------------------------------------------------------------
__global__ __launch_bounds__(128)
void gru_decoder_kernel(const float* __restrict__ x,
                        const float* __restrict__ noise_z,
                        const float* __restrict__ context,
                        const float* __restrict__ bi0, const float* __restrict__ bh0,
                        const float* __restrict__ bi1, const float* __restrict__ bh1,
                        const float* __restrict__ ba,
                        const bf16_t* __restrict__ ws,
                        float* __restrict__ out) {
  extern __shared__ __align__(16) char smem[];
  bf16_t* sWi1 = (bf16_t*)smem;           // 49152 elems (96 KB)
  bf16_t* sWh1 = sWi1 + 49152;            // 49152 elems (96 KB)
  bf16_t* sWi0 = sWh1 + 49152;            // 12288 elems (24 KB)
  bf16_t* sWap = sWi0 + 12288;            //  2048 elems ( 4 KB)
  float*  sB   = (float*)(sWap + 2048);   //  1032 floats
  bf16_t* sAct = (bf16_t*)(sB + 1032);    // per wave: X(512)+H0(2048)+H1(2048)

  const int tid  = threadIdx.x;
  const int wave = tid >> 5;
  const int lane = tid & 31;
  const int idx  = lane & 15;
  const int hlf  = lane >> 4;
  const int rowbase = blockIdx.x * (WAVES * RPW) + wave * RPW;

  bf16_t* X  = sAct + wave * 4608;  // [16][32]  layer-0 input (cols 12..31 zero)
  bf16_t* H0 = X + 512;             // [16][128] bf16 hidden, layer 0
  bf16_t* H1 = H0 + 2048;           // [16][128] bf16 hidden, layer 1

  // ---- cooperative init: weights (contiguous 225280 B), biases, state ----
  {
    const uint4* src = (const uint4*)(ws + WS_WI1);
    uint4* dst = (uint4*)sWi1;
    for (int i = tid; i < 14080; i += 128) dst[i] = src[i];
  }
  for (int i = tid; i < 256; i += 128) {
    sB[i]       = bi0[i] + bh0[i];
    sB[512 + i] = bi1[i] + bh1[i];
  }
  for (int i = tid; i < 128; i += 128) {
    sB[256 + i] = bi0[256 + i];  sB[384 + i] = bh0[256 + i];
    sB[768 + i] = bi1[256 + i];  sB[896 + i] = bh1[256 + i];
  }
  if (tid < SDIM) sB[1024 + tid] = ba[tid];

  for (int i = lane; i < RPW * HD; i += 32) {
    H0[i] = (bf16_t)0.0f;
    H1[i] = (bf16_t)0.0f;
  }
  for (int i = lane; i < RPW * 32; i += 32) {
    int m = i >> 5, c = i & 31;
    float v = 0.0f;
    if (c < SDIM)             v = noise_z[(rowbase + m) * SDIM + c];
    else if (c < SDIM + EDIM) v = context[(rowbase + m) * EDIM + (c - SDIM)];
    X[i] = (bf16_t)v;
  }
  __syncthreads();

  const bf16_t* gWh0 = ws + WS_WH0;  // layer-0 recurrent weights stay in L2/WGP$
  v16bf aH[4], aI1[4], aX[1];

  auto two_layer = [&]() {
#pragma unroll
    for (int k = 0; k < 4; ++k) aH[k] = fragLd(H0, idx, HD, k * 32, hlf);
    aX[0] = fragLd(X, idx, 32, 0, hlf);
    gru_layer<1, 32>(idx, hlf, aH, aX, gWh0, sWi0, sB, H0);
#pragma unroll
    for (int k = 0; k < 4; ++k) {
      aI1[k] = fragLd(H0, idx, HD, k * 32, hlf);  // new h0 = layer-1 input
      aH[k]  = fragLd(H1, idx, HD, k * 32, hlf);  // old h1
    }
    gru_layer<4, HD>(idx, hlf, aH, aI1, sWh1, sWi1, sB + 512, H1);
  };

  // begin_state: one step on cat(noise_z, context) from zero hidden state
  two_layer();

  // first decoder input = x[:, 0]
  {
    int m = lane >> 1, c = lane & 1;
    X[m * 32 + c] = (bf16_t)x[((size_t)(rowbase + m) * TLEN + 0) * SDIM + c];
  }

#pragma unroll 1
  for (int t = 0; t < TLEN; ++t) {
    two_layer();
    // head: out = h1 @ Wa^T + ba  (one padded 16-col WMMA tile)
    const float bav = (idx < SDIM) ? sB[1024 + idx] : 0.0f;
    v8f co = {bav, bav, bav, bav, bav, bav, bav, bav};
#pragma unroll
    for (int k = 0; k < 4; ++k) {
      v16bf aO = fragLd(H1, idx, HD, k * 32, hlf);
      co = wmma_bf16(aO, fragLd(sWap, idx, HD, k * 32, hlf), co);
    }
    if (idx < SDIM) {
#pragma unroll
      for (int v = 0; v < 8; ++v) {
        const int m = v + hlf * 8;
        const float o = co[v];
        out[((size_t)(rowbase + m) * TLEN + t) * SDIM + idx] = o;
        X[m * 32 + idx] = (bf16_t)o;  // autoregressive feedback
      }
    }
  }
}

extern "C" void kernel_launch(void* const* d_in, const int* in_sizes, int n_in,
                              void* d_out, int out_size, void* d_ws, size_t ws_size,
                              hipStream_t stream) {
  (void)in_sizes; (void)n_in; (void)out_size; (void)ws_size;
  const float* x   = (const float*)d_in[0];
  const float* nz  = (const float*)d_in[1];
  const float* ctx = (const float*)d_in[2];
  const float* Wi0 = (const float*)d_in[3];
  const float* Wh0 = (const float*)d_in[4];
  const float* bi0 = (const float*)d_in[5];
  const float* bh0 = (const float*)d_in[6];
  const float* Wi1 = (const float*)d_in[7];
  const float* Wh1 = (const float*)d_in[8];
  const float* bi1 = (const float*)d_in[9];
  const float* bh1 = (const float*)d_in[10];
  const float* Wa  = (const float*)d_in[11];
  const float* ba  = (const float*)d_in[12];
  bf16_t* ws = (bf16_t*)d_ws;
  float* out = (float*)d_out;

  prep_kernel<<<(WS_TOTAL + 255) / 256, 256, 0, stream>>>(Wh0, Wi1, Wh1, Wi0, Wa, ws);

  // dynamic LDS: 192K (Wi1+Wh1) + 24K (Wi0 pad) + 4K (Wa pad) + biases + 4 wave bufs
  size_t smem = (size_t)49152 * 2 * 2 + 12288 * 2 + 2048 * 2 + 1032 * 4 +
                (size_t)WAVES * 4608 * 2;  // = 266272 bytes (< 320 KB/WGP)
  gru_decoder_kernel<<<NBLK, 128, smem, stream>>>(x, nz, ctx, bi0, bh0, bi1, bh1,
                                                  ba, ws, out);
}
